// AttentionBase_90718299226474
// MI455X (gfx1250) — matrix operand order repaired and checked by reading, not compiled, not run on hardware
//
#include <hip/hip_runtime.h>

typedef __attribute__((ext_vector_type(16))) _Float16 v16h;
typedef __attribute__((ext_vector_type(8)))  float    v8f;

#define D_MODEL 1024
#define N_TOK   2048
#define BATCH   2
#define NHEAD   16
#define HDIM    64
#define M_TOTAL (BATCH * N_TOK)   /* 4096 */

// ---------------------------------------------------------------------------
// Fragment loaders per CDNA5 ISA 7.12.2 VGPR layouts (16-bit A 16x32, B 32x16)
// ---------------------------------------------------------------------------
union AF { v16h h; unsigned int u[8]; };

// A-matrix 16x32 f16: lane row = lane%16; pair i -> K = koff + 2i + (i<4?0:8),
// koff = (lane<16 ? 0 : 8). base points at (row0, k0); stride in elements (even).
__device__ __forceinline__ v16h load_a_frag(const _Float16* base, int stride) {
  const int lane = threadIdx.x & 31;
  const int row  = lane & 15;
  const int koff = (lane < 16) ? 0 : 8;
  AF f;
#pragma unroll
  for (int i = 0; i < 8; ++i) {
    const int k = koff + 2 * i + ((i < 4) ? 0 : 8);
    f.u[i] = *reinterpret_cast<const unsigned int*>(base + row * stride + k);
  }
  return f.h;
}

// B-matrix 32x16 f16 stored [N][K] in LDS: lane col n = lane%16,
// pair i -> K = kst + 2i + (i<4?0:8), kst = (lane<16 ? 0 : 8).
__device__ __forceinline__ v16h load_b_frag(const _Float16* base, int stride) {
  const int lane = threadIdx.x & 31;
  const int n    = lane & 15;
  const int kst  = (lane < 16) ? 0 : 8;
  AF f;
#pragma unroll
  for (int i = 0; i < 8; ++i)
    f.u[i] = *reinterpret_cast<const unsigned int*>(base + n * stride + kst + 2 * i + ((i < 4) ? 0 : 8));
  return f.h;
}

// Async global->LDS copy (CDNA5, ASYNCcnt-tracked, no VGPR data movement)
__device__ __forceinline__ void async_copy_b128(const _Float16* gsrc, _Float16* ldst) {
  const unsigned lds = (unsigned)(size_t)ldst;
  asm volatile("global_load_async_to_lds_b128 %0, %1, off"
               :: "v"(lds), "v"(gsrc) : "memory");
}
__device__ __forceinline__ void wait_async0() {
  asm volatile("s_wait_asynccnt 0" ::: "memory");
}

// ---------------------------------------------------------------------------
// Kernel 1: QKV projection  y = x @ W  (fp32 in, f16 head-interleaved out)
// block 256 (8 waves), tile 128x128, K-chunk 32; each wave owns 32x64
// ---------------------------------------------------------------------------
#define AS 40
#define BS 36

__global__ __launch_bounds__(256) void proj_gemm_kernel(
    const float* __restrict__ A,      // [4096,1024] fp32
    const float* __restrict__ W,      // [1024,1024] fp32
    _Float16* __restrict__ dst)       // [B,H,N,HDIM] f16
{
  __shared__ _Float16 As[128 * AS];
  __shared__ _Float16 Bt[128 * BS];
  const int t    = threadIdx.x;
  const int m0   = blockIdx.x * 128;
  const int n0   = blockIdx.y * 128;
  const int wid  = t >> 5;
  const int lane = t & 31;
  const int wm   = wid >> 1;   // 0..3 -> 32 rows each
  const int wn   = wid & 1;    // 0..1 -> 64 cols each

  v8f zero = {};
  v8f acc[2][4];
#pragma unroll
  for (int mi = 0; mi < 2; ++mi)
#pragma unroll
    for (int ni = 0; ni < 4; ++ni) acc[mi][ni] = zero;

  for (int kb = 0; kb < D_MODEL; kb += 32) {
    __syncthreads();
    // stage A tile 128x32 fp32 -> f16 (conversion: must go through VALU)
#pragma unroll
    for (int i = 0; i < 4; ++i) {
      const int idx = t + i * 256;       // 0..1023 float4 slots
      const int row = idx >> 3;
      const int c4  = (idx & 7) << 2;
      const float4 v = *reinterpret_cast<const float4*>(
          &A[(size_t)(m0 + row) * D_MODEL + kb + c4]);
      _Float16* p = &As[row * AS + c4];
      p[0] = (_Float16)v.x; p[1] = (_Float16)v.y;
      p[2] = (_Float16)v.z; p[3] = (_Float16)v.w;
    }
    // stage W tile transposed: Bt[n][k] = W[kb+k][n0+n], 32x128
#pragma unroll
    for (int i = 0; i < 4; ++i) {
      const int idx = t + i * 256;       // 0..1023 float4 slots
      const int k   = idx >> 5;          // 0..31
      const int n4  = (idx & 31) << 2;   // 0..124
      const float4 v = *reinterpret_cast<const float4*>(
          &W[(size_t)(kb + k) * D_MODEL + n0 + n4]);
      Bt[(n4 + 0) * BS + k] = (_Float16)v.x;
      Bt[(n4 + 1) * BS + k] = (_Float16)v.y;
      Bt[(n4 + 2) * BS + k] = (_Float16)v.z;
      Bt[(n4 + 3) * BS + k] = (_Float16)v.w;
    }
    __syncthreads();

    const v16h a0 = load_a_frag(&As[(wm * 32 + 0)  * AS], AS);
    const v16h a1 = load_a_frag(&As[(wm * 32 + 16) * AS], AS);
    v16h bfr[4];
#pragma unroll
    for (int ni = 0; ni < 4; ++ni)
      bfr[ni] = load_b_frag(&Bt[(wn * 64 + ni * 16) * BS], BS);
#pragma unroll
    for (int ni = 0; ni < 4; ++ni) {
      acc[0][ni] = __builtin_amdgcn_wmma_f32_16x16x32_f16(false, a0, false, bfr[ni], (short)0, acc[0][ni], false, false);
      acc[1][ni] = __builtin_amdgcn_wmma_f32_16x16x32_f16(false, a1, false, bfr[ni], (short)0, acc[1][ni], false, false);
    }
  }

  // writeback, head-interleaved: dst[b][h][tok][d]
#pragma unroll
  for (int mi = 0; mi < 2; ++mi)
#pragma unroll
    for (int ni = 0; ni < 4; ++ni)
#pragma unroll
      for (int r = 0; r < 8; ++r) {
        const int gm  = m0 + wm * 32 + mi * 16 + r + ((lane < 16) ? 0 : 8);
        const int gn  = n0 + wn * 64 + ni * 16 + (lane & 15);
        const int b   = gm >> 11;
        const int tok = gm & 2047;
        const int h   = gn >> 6;
        const int d   = gn & 63;
        dst[(((size_t)(b * NHEAD + h)) * N_TOK + tok) * HDIM + d] =
            (_Float16)acc[mi][ni][r];
      }
}

// ---------------------------------------------------------------------------
// Kernel 2: fused flash attention per (b,h,64-row Q chunk); 4 waves
// ---------------------------------------------------------------------------
#define QS 72
#define KS 72
#define VS 72
#define PS 72

__global__ __launch_bounds__(128) void attn_kernel(
    const _Float16* __restrict__ Q,   // [B,H,N,HDIM]
    const _Float16* __restrict__ K,
    const _Float16* __restrict__ V,
    _Float16* __restrict__ ctx)       // [B,N,D] merged heads
{
  __shared__ _Float16 Qs[64 * QS];
  __shared__ _Float16 Ks[64 * KS];
  __shared__ _Float16 Vt[64 * VS];    // transposed: [hd][key]
  __shared__ _Float16 Ps[64 * PS];    // per-wave 16x64 P tiles

  const int t    = threadIdx.x;
  const int lane = t & 31;
  const int w    = t >> 5;            // 0..3
  const int q0   = blockIdx.x * 64;
  const int h    = blockIdx.y;
  const int b    = blockIdx.z;
  const size_t headoff = ((size_t)(b * NHEAD + h)) * N_TOK * HDIM;
  const _Float16* Qh = Q + headoff;
  const _Float16* Kh = K + headoff;
  const _Float16* Vh = V + headoff;

  // stage Q chunk 64x64 via async B128 copies (ASYNCcnt path)
#pragma unroll
  for (int i = 0; i < 4; ++i) {
    const int idx = t + i * 128;        // 0..511 (8-half slots)
    const int r   = idx >> 3;
    const int d8  = (idx & 7) << 3;
    async_copy_b128(&Qh[(size_t)(q0 + r) * HDIM + d8], &Qs[r * QS + d8]);
  }
  wait_async0();
  __syncthreads();

  const v16h aq0 = load_a_frag(&Qs[(w * 16) * QS + 0],  QS);
  const v16h aq1 = load_a_frag(&Qs[(w * 16) * QS + 32], QS);

  v8f zero = {};
  v8f o[4];
  o[0] = zero; o[1] = zero; o[2] = zero; o[3] = zero;
  float m_i[8], l_i[8];
#pragma unroll
  for (int r = 0; r < 8; ++r) { m_i[r] = -1e30f; l_i[r] = 0.0f; }

  for (int j = 0; j < N_TOK / 64; ++j) {
    __syncthreads();
    // K chunk: pure byte copy -> async to LDS
#pragma unroll
    for (int i = 0; i < 4; ++i) {
      const int idx = t + i * 128;
      const int r   = idx >> 3;
      const int d8  = (idx & 7) << 3;
      async_copy_b128(&Kh[(size_t)(j * 64 + r) * HDIM + d8], &Ks[r * KS + d8]);
    }
    // V chunk: needs transpose -> regular loads + scalar LDS stores
#pragma unroll
    for (int i = 0; i < 4; ++i) {
      const int idx = t + i * 128;
      const int r   = idx >> 3;
      const int d8  = (idx & 7) << 3;
      union { uint4 v; _Float16 hh[8]; } uv;
      uv.v = *reinterpret_cast<const uint4*>(&Vh[(size_t)(j * 64 + r) * HDIM + d8]);
#pragma unroll
      for (int q = 0; q < 8; ++q)
        Vt[(d8 + q) * VS + r] = uv.hh[q];
    }
    wait_async0();
    __syncthreads();

    // S = (Q K^T) * D^-0.5 for this wave's 16 rows, 64 keys
    v8f s[4];
#pragma unroll
    for (int nt = 0; nt < 4; ++nt) {
      v8f sv = zero;
      const v16h bk0 = load_b_frag(&Ks[(nt * 16) * KS + 0],  KS);
      const v16h bk1 = load_b_frag(&Ks[(nt * 16) * KS + 32], KS);
      sv = __builtin_amdgcn_wmma_f32_16x16x32_f16(false, aq0, false, bk0, (short)0, sv, false, false);
      sv = __builtin_amdgcn_wmma_f32_16x16x32_f16(false, aq1, false, bk1, (short)0, sv, false, false);
      s[nt] = sv;
    }

    const float sc = 0.03125f;          // 1024^-0.5 (full model dim, per reference)
    float rm[8];
#pragma unroll
    for (int r = 0; r < 8; ++r) {
      float v = -1e30f;
#pragma unroll
      for (int nt = 0; nt < 4; ++nt) { s[nt][r] *= sc; v = fmaxf(v, s[nt][r]); }
#pragma unroll
      for (int msk = 1; msk < 16; msk <<= 1)
        v = fmaxf(v, __shfl_xor(v, msk, 32));   // halves stay separate (masks < 16)
      rm[r] = v;
    }

#pragma unroll
    for (int r = 0; r < 8; ++r) {
      const float mn    = fmaxf(m_i[r], rm[r]);
      const float alpha = __expf(m_i[r] - mn);
      float rs = 0.0f;
#pragma unroll
      for (int nt = 0; nt < 4; ++nt) {
        const float p = __expf(s[nt][r] - mn);
        rs += p;
        Ps[(w * 16 + r + ((lane < 16) ? 0 : 8)) * PS + nt * 16 + (lane & 15)] = (_Float16)p;
      }
#pragma unroll
      for (int msk = 1; msk < 16; msk <<= 1) rs += __shfl_xor(rs, msk, 32);
      l_i[r] = l_i[r] * alpha + rs;
      m_i[r] = mn;
      o[0][r] *= alpha; o[1][r] *= alpha; o[2][r] *= alpha; o[3][r] *= alpha;
    }

    // wave-local LDS RAW fence (CDNA5 split DS counter) before P-frag reloads
    asm volatile("s_wait_dscnt 0" ::: "memory");

    // O += P @ V
#pragma unroll
    for (int kk = 0; kk < 2; ++kk) {
      const v16h ap = load_a_frag(&Ps[(w * 16) * PS + kk * 32], PS);
#pragma unroll
      for (int nt = 0; nt < 4; ++nt) {
        const v16h bv = load_b_frag(&Vt[(nt * 16) * VS + kk * 32], VS);
        o[nt] = __builtin_amdgcn_wmma_f32_16x16x32_f16(false, ap, false, bv, (short)0, o[nt], false, false);
      }
    }
  }

  // finalize: O / l, write merged-head context f16
#pragma unroll
  for (int r = 0; r < 8; ++r) {
    const float inv = 1.0f / l_i[r];
    const int tok = q0 + w * 16 + r + ((lane < 16) ? 0 : 8);
#pragma unroll
    for (int nt = 0; nt < 4; ++nt) {
      const int d = nt * 16 + (lane & 15);
      ctx[((size_t)(b * N_TOK) + tok) * D_MODEL + h * HDIM + d] =
          (_Float16)(o[nt][r] * inv);
    }
  }
}

// ---------------------------------------------------------------------------
// Kernel 3: output projection  out = ctx @ W_o  (f16 A, fp32 out)
// block 256 (8 waves), tile 128x128
// ---------------------------------------------------------------------------
__global__ __launch_bounds__(256) void out_gemm_kernel(
    const _Float16* __restrict__ A,   // [4096,1024] f16
    const float* __restrict__ W,      // [1024,1024] fp32
    float* __restrict__ out)          // [4096,1024] fp32
{
  __shared__ _Float16 As[128 * AS];
  __shared__ _Float16 Bt[128 * BS];
  const int t    = threadIdx.x;
  const int m0   = blockIdx.x * 128;
  const int n0   = blockIdx.y * 128;
  const int wid  = t >> 5;
  const int lane = t & 31;
  const int wm   = wid >> 1;
  const int wn   = wid & 1;

  v8f zero = {};
  v8f acc[2][4];
#pragma unroll
  for (int mi = 0; mi < 2; ++mi)
#pragma unroll
    for (int ni = 0; ni < 4; ++ni) acc[mi][ni] = zero;

  for (int kb = 0; kb < D_MODEL; kb += 32) {
    __syncthreads();
    // stage A tile 128x32 f16: pure byte copy -> async B128 to LDS
#pragma unroll
    for (int i = 0; i < 2; ++i) {
      const int idx = t + i * 256;       // 0..511 (8-half slots)
      const int row = idx >> 2;
      const int c8  = (idx & 3) << 3;
      async_copy_b128(&A[(size_t)(m0 + row) * D_MODEL + kb + c8],
                      &As[row * AS + c8]);     // AS=40 -> 16B-aligned LDS dest
    }
    // stage W tile transposed (fp32 -> f16 conversion on VALU)
#pragma unroll
    for (int i = 0; i < 4; ++i) {
      const int idx = t + i * 256;
      const int k   = idx >> 5;
      const int n4  = (idx & 31) << 2;
      const float4 v = *reinterpret_cast<const float4*>(
          &W[(size_t)(kb + k) * D_MODEL + n0 + n4]);
      Bt[(n4 + 0) * BS + k] = (_Float16)v.x;
      Bt[(n4 + 1) * BS + k] = (_Float16)v.y;
      Bt[(n4 + 2) * BS + k] = (_Float16)v.z;
      Bt[(n4 + 3) * BS + k] = (_Float16)v.w;
    }
    wait_async0();
    __syncthreads();

    const v16h a0 = load_a_frag(&As[(wm * 32 + 0)  * AS], AS);
    const v16h a1 = load_a_frag(&As[(wm * 32 + 16) * AS], AS);
    v16h bfr[4];
#pragma unroll
    for (int ni = 0; ni < 4; ++ni)
      bfr[ni] = load_b_frag(&Bt[(wn * 64 + ni * 16) * BS], BS);
#pragma unroll
    for (int ni = 0; ni < 4; ++ni) {
      acc[0][ni] = __builtin_amdgcn_wmma_f32_16x16x32_f16(false, a0, false, bfr[ni], (short)0, acc[0][ni], false, false);
      acc[1][ni] = __builtin_amdgcn_wmma_f32_16x16x32_f16(false, a1, false, bfr[ni], (short)0, acc[1][ni], false, false);
    }
  }

#pragma unroll
  for (int mi = 0; mi < 2; ++mi)
#pragma unroll
    for (int ni = 0; ni < 4; ++ni)
#pragma unroll
      for (int r = 0; r < 8; ++r) {
        const int gm = m0 + wm * 32 + mi * 16 + r + ((lane < 16) ? 0 : 8);
        const int gn = n0 + wn * 64 + ni * 16 + (lane & 15);
        out[(size_t)gm * D_MODEL + gn] = acc[mi][ni][r];
      }
}

// ---------------------------------------------------------------------------
extern "C" void kernel_launch(void* const* d_in, const int* in_sizes, int n_in,
                              void* d_out, int out_size, void* d_ws, size_t ws_size,
                              hipStream_t stream) {
  const float* x  = (const float*)d_in[0];
  const float* Wq = (const float*)d_in[1];
  const float* Wk = (const float*)d_in[2];
  const float* Wv = (const float*)d_in[3];
  const float* Wo = (const float*)d_in[4];
  float* out = (float*)d_out;

  const size_t elems = (size_t)M_TOTAL * D_MODEL;   // 4M f16 each
  _Float16* q   = (_Float16*)d_ws;
  _Float16* k   = q + elems;
  _Float16* v   = k + elems;
  _Float16* ctx = v + elems;

  dim3 gg(M_TOTAL / 128, D_MODEL / 128);
  proj_gemm_kernel<<<gg, 256, 0, stream>>>(x, Wq, q);
  proj_gemm_kernel<<<gg, 256, 0, stream>>>(x, Wk, k);
  proj_gemm_kernel<<<gg, 256, 0, stream>>>(x, Wv, v);

  dim3 ga(N_TOK / 64, NHEAD, BATCH);
  attn_kernel<<<ga, 128, 0, stream>>>(q, k, v, ctx);

  out_gemm_kernel<<<gg, 256, 0, stream>>>(ctx, Wo, out);
}
